// LocalAgg_39324720562662
// MI455X (gfx1250) — matrix operand order repaired and checked by reading, not compiled, not run on hardware
//
#include <hip/hip_runtime.h>
#include <hip/hip_bf16.h>
#include <math.h>

// ---------------------------------------------------------------------------
// LocalAgg on MI455X (gfx1250): per-center 32x35 -> 32x64 -> 32x64 MLP via
// v_wmma_f32_16x16x32_f16, max-pool over K=32 neighbors, fused skip GEMM.
// One wave32 per center; weights staged (transposed) in LDS once per block.
// Round 3: hoist A-fragments out of the column-tile loop (compiler was
// re-loading them from LDS per nt due to conservative aliasing vs sH1 stores).
// ---------------------------------------------------------------------------

typedef __attribute__((ext_vector_type(16))) _Float16 v16h;
typedef __attribute__((ext_vector_type(8)))  _Float16 v8h;
typedef __attribute__((ext_vector_type(8)))  float    v8f;
typedef __attribute__((ext_vector_type(4)))  float    v4f;

union V16 { v16h v; v8h h[2]; };

static constexpr int Bb   = 16;
static constexpr int Nn   = 8192;
static constexpr int Mm   = 2048;
static constexpr int Kk   = 32;    // neighbors per center
static constexpr int CIN  = 32;
static constexpr int COUT = 64;
static constexpr int LIN  = 35;    // 3 + CIN, padded to 64 for WMMA
static constexpr int WAVES = 4;    // waves per block (128 threads)

// Branchless exact-GELU: erf via Abramowitz-Stegun 7.1.26 (|err|<=1.5e-7),
// hardware v_rcp_f32 + v_exp_f32, sign restored with copysign (v_bfi).
__device__ __forceinline__ float gelu_fast(float x) {
  const float z  = x * 0.7071067811865475f;          // x / sqrt(2)
  const float az = __builtin_fabsf(z);
  const float t  = __builtin_amdgcn_rcpf(__builtin_fmaf(0.3275911f, az, 1.0f));
  float p = __builtin_fmaf(1.061405429f, t, -1.453152027f);
  p = __builtin_fmaf(p, t, 1.421413741f);
  p = __builtin_fmaf(p, t, -0.284496736f);
  p = __builtin_fmaf(p, t, 0.254829592f);
  p *= t;
  const float e = __builtin_amdgcn_exp2f(az * az * -1.4426950408889634f); // e^{-z^2}
  const float erf_abs = __builtin_fmaf(-p, e, 1.0f);
  const float erf_z   = __builtin_copysignf(erf_abs, z);
  return 0.5f * x * (1.0f + erf_z);
}

// A-fragment (16-bit A 16x32 per ISA layout): lane L -> row (L&15)+16*rt,
// halves 0..7 = K k0..k0+7, halves 8..15 = K k0+16..k0+23, k0 = kt*32 + (hi?8:0)
__device__ __forceinline__ v16h load_afrag(const _Float16* base /*32x64 row-major*/,
                                           int rt, int kt, int lane) {
  const int row = rt * 16 + (lane & 15);
  const int k0  = kt * 32 + ((lane & 16) ? 8 : 0);
  const _Float16* p = base + row * 64 + k0;
  V16 u;
  u.h[0] = *(const v8h*)(p);
  u.h[1] = *(const v8h*)(p + 16);
  return u.v;
}

// B-fragment (16-bit B 32x16): lane L -> col (L&15)+16*nt,
// halves 0..7 = K k0..k0+7, halves 8..15 = K k0+16..k0+23, k0 = kt*32 + (hi?8:0).  wT is [n][k] (64x64).
__device__ __forceinline__ v16h load_bfrag(const _Float16* wT,
                                           int kt, int nt, int lane) {
  const int n  = nt * 16 + (lane & 15);
  const int k0 = kt * 32 + ((lane & 16) ? 8 : 0);
  const _Float16* p = wT + n * 64 + k0;
  V16 u;
  u.h[0] = *(const v8h*)(p);
  u.h[1] = *(const v8h*)(p + 16);
  return u.v;
}

__global__ __launch_bounds__(WAVES * 32, 1)
void localagg_wmma_kernel(const float* __restrict__ feat,  const float* __restrict__ pos,
                          const float* __restrict__ cfeat, const float* __restrict__ cpos,
                          const int*   __restrict__ knn,
                          const float* __restrict__ W1, const float* __restrict__ b1,
                          const float* __restrict__ g1, const float* __restrict__ be1,
                          const float* __restrict__ m1, const float* __restrict__ v1,
                          const float* __restrict__ W2, const float* __restrict__ b2,
                          const float* __restrict__ g2, const float* __restrict__ be2,
                          const float* __restrict__ m2, const float* __restrict__ v2,
                          const float* __restrict__ Wskip,
                          float* __restrict__ out) {
  __shared__ _Float16 sW1t[64 * 64];                 // W1^T padded: [n][k]
  __shared__ _Float16 sW2t[64 * 64];                 // W2^T: [n][k]
  __shared__ float    sWskip[CIN * COUT];            // [k][n]
  __shared__ float    sScale1[64], sShift1[64], sScale2[64], sShift2[64];
  __shared__ _Float16 sIn[WAVES][32 * 64];           // per-wave local_in, row-major
  __shared__ _Float16 sH1[WAVES][32 * 64];           // per-wave hidden, row-major
  __shared__ float    sAgg[WAVES][64];

  const int tid = threadIdx.x;

  // ---- stage weights (transposed, zero-padded) + fused BN scale/shift ----
  for (int i = tid; i < 64 * 64; i += WAVES * 32) {
    const int n = i >> 6, k = i & 63;
    sW1t[i] = (k < LIN) ? (_Float16)W1[k * 64 + n] : (_Float16)0.0f;
    sW2t[i] = (_Float16)W2[k * 64 + n];
  }
  for (int i = tid; i < CIN * COUT; i += WAVES * 32) sWskip[i] = Wskip[i];
  if (tid < 64) {
    const float s1 = g1[tid] * rsqrtf(v1[tid] + 1e-5f);
    sScale1[tid] = s1;
    sShift1[tid] = (b1[tid] - m1[tid]) * s1 + be1[tid];
    const float s2 = g2[tid] * rsqrtf(v2[tid] + 1e-5f);
    sScale2[tid] = s2;
    sShift2[tid] = (b2[tid] - m2[tid]) * s2 + be2[tid];
  }
  __syncthreads();

  const int lane   = tid & 31;
  const int wv     = tid >> 5;
  const int waveId = blockIdx.x * WAVES + wv;
  const int nWaves = gridDim.x * WAVES;
  const int total  = Bb * Mm;

  _Float16* myIn = &sIn[wv][0];
  _Float16* myH1 = &sH1[wv][0];
  const int col_lo = lane & 15;
  const int rsel   = (lane & 16) ? 8 : 0;

  for (int c = waveId; c < total; c += nWaves) {
    const int b = c / Mm;

    // prefetch next iteration's knn line (global_prefetch_b8)
    const int cn = c + nWaves;
    if (cn < total) __builtin_prefetch(knn + (size_t)cn * Kk + lane, 0, 0);

    // ---- gather one neighbor row per lane, build [delta_xyz | feat | 0pad] ----
    {
      const int nbr = knn[(size_t)c * Kk + lane];
      const float* fp = feat + ((size_t)b * Nn + nbr) * CIN;
      const float* pp = pos  + ((size_t)b * Nn + nbr) * 3;
      const float* cp = cpos + (size_t)c * 3;
      union Row { _Float16 e[64]; v8h v[8]; } rw;
      rw.e[0] = (_Float16)(pp[0] - cp[0]);
      rw.e[1] = (_Float16)(pp[1] - cp[1]);
      rw.e[2] = (_Float16)(pp[2] - cp[2]);
      const v4f* f4 = (const v4f*)fp;
#pragma unroll
      for (int q = 0; q < 8; q++) {
        const v4f fv = f4[q];
        rw.e[3 + 4 * q + 0] = (_Float16)fv.x;
        rw.e[3 + 4 * q + 1] = (_Float16)fv.y;
        rw.e[3 + 4 * q + 2] = (_Float16)fv.z;
        rw.e[3 + 4 * q + 3] = (_Float16)fv.w;
      }
#pragma unroll
      for (int q = LIN; q < 64; q++) rw.e[q] = (_Float16)0.0f;
      v8h* dst = (v8h*)(myIn + lane * 64);
#pragma unroll
      for (int q = 0; q < 8; q++) dst[q] = rw.v[q];
    }

    // ---- GEMM1: (32x64) @ (64x64) -> BN+GELU -> sH1 ----
    {
      // A-fragments are nt-invariant: load once, hold in VGPRs.
      const v16h a00 = load_afrag(myIn, 0, 0, lane);
      const v16h a01 = load_afrag(myIn, 0, 1, lane);
      const v16h a10 = load_afrag(myIn, 1, 0, lane);
      const v16h a11 = load_afrag(myIn, 1, 1, lane);
#pragma unroll
      for (int nt = 0; nt < 4; nt++) {
        const v16h bf0 = load_bfrag(sW1t, 0, nt, lane);
        const v16h bf1 = load_bfrag(sW1t, 1, nt, lane);
        v8f acc0 = {}; v8f acc1 = {};
        acc0 = __builtin_amdgcn_wmma_f32_16x16x32_f16(false, a00, false, bf0,
                                                      (short)0, acc0, false, false);
        acc0 = __builtin_amdgcn_wmma_f32_16x16x32_f16(false, a01, false, bf1,
                                                      (short)0, acc0, false, false);
        acc1 = __builtin_amdgcn_wmma_f32_16x16x32_f16(false, a10, false, bf0,
                                                      (short)0, acc1, false, false);
        acc1 = __builtin_amdgcn_wmma_f32_16x16x32_f16(false, a11, false, bf1,
                                                      (short)0, acc1, false, false);
        const int col = nt * 16 + col_lo;
        const float s = sScale1[col], t = sShift1[col];
#pragma unroll
        for (int r = 0; r < 8; r++) {
          const float e0 = gelu_fast(__builtin_fmaf(acc0[r], s, t));
          const float e1 = gelu_fast(__builtin_fmaf(acc1[r], s, t));
          myH1[(r + rsel) * 64 + col]      = (_Float16)e0;
          myH1[(16 + r + rsel) * 64 + col] = (_Float16)e1;
        }
      }
    }

    // ---- GEMM2: (32x64) @ (64x64) -> BN+GELU -> max over 32 rows ----
    {
      const v16h a00 = load_afrag(myH1, 0, 0, lane);
      const v16h a01 = load_afrag(myH1, 0, 1, lane);
      const v16h a10 = load_afrag(myH1, 1, 0, lane);
      const v16h a11 = load_afrag(myH1, 1, 1, lane);
#pragma unroll
      for (int nt = 0; nt < 4; nt++) {
        const v16h bf0 = load_bfrag(sW2t, 0, nt, lane);
        const v16h bf1 = load_bfrag(sW2t, 1, nt, lane);
        v8f acc0 = {}; v8f acc1 = {};
        acc0 = __builtin_amdgcn_wmma_f32_16x16x32_f16(false, a00, false, bf0,
                                                      (short)0, acc0, false, false);
        acc0 = __builtin_amdgcn_wmma_f32_16x16x32_f16(false, a01, false, bf1,
                                                      (short)0, acc0, false, false);
        acc1 = __builtin_amdgcn_wmma_f32_16x16x32_f16(false, a10, false, bf0,
                                                      (short)0, acc1, false, false);
        acc1 = __builtin_amdgcn_wmma_f32_16x16x32_f16(false, a11, false, bf1,
                                                      (short)0, acc1, false, false);
        const int col = nt * 16 + col_lo;
        const float s = sScale2[col], t = sShift2[col];
        float mx = -3.4e38f;
#pragma unroll
        for (int r = 0; r < 8; r++) {
          mx = fmaxf(mx, gelu_fast(__builtin_fmaf(acc0[r], s, t)));
          mx = fmaxf(mx, gelu_fast(__builtin_fmaf(acc1[r], s, t)));
        }
        // lanes L and L^16 hold the same column; combine their row subsets
        mx = fmaxf(mx, __shfl_xor(mx, 16, 32));
        sAgg[wv][col] = mx;
      }
    }

    // ---- skip GEMM (32 -> 64) + write out (each lane: cols lane, lane+32) ----
    const float cfv = cfeat[(size_t)c * CIN + lane];
    float s0 = 0.0f, s1 = 0.0f;
#pragma unroll
    for (int k2 = 0; k2 < CIN; k2++) {
      const float bc = __shfl(cfv, k2, 32);
      s0 = fmaf(bc, sWskip[k2 * COUT + lane], s0);
      s1 = fmaf(bc, sWskip[k2 * COUT + 32 + lane], s1);
    }
    out[(size_t)c * COUT + lane]      = sAgg[wv][lane] + s0;
    out[(size_t)c * COUT + 32 + lane] = sAgg[wv][32 + lane] + s1;
  }
}

extern "C" void kernel_launch(void* const* d_in, const int* in_sizes, int n_in,
                              void* d_out, int out_size, void* d_ws, size_t ws_size,
                              hipStream_t stream) {
  const float* feat  = (const float*)d_in[0];
  const float* pos   = (const float*)d_in[1];
  const float* cfeat = (const float*)d_in[2];
  const float* cpos  = (const float*)d_in[3];
  const int*   knn   = (const int*)d_in[4];
  const float* W1    = (const float*)d_in[5];
  const float* b1    = (const float*)d_in[6];
  const float* g1    = (const float*)d_in[7];
  const float* be1   = (const float*)d_in[8];
  const float* m1    = (const float*)d_in[9];
  const float* v1    = (const float*)d_in[10];
  const float* W2    = (const float*)d_in[11];
  const float* b2    = (const float*)d_in[12];
  const float* g2    = (const float*)d_in[13];
  const float* be2   = (const float*)d_in[14];
  const float* m2    = (const float*)d_in[15];
  const float* v2    = (const float*)d_in[16];
  const float* Wskip = (const float*)d_in[17];
  float* out = (float*)d_out;

  dim3 grid(1024), block(WAVES * 32);
  hipLaunchKernelGGL(localagg_wmma_kernel, grid, block, 0, stream,
                     feat, pos, cfeat, cpos, knn,
                     W1, b1, g1, be1, m1, v1,
                     W2, b2, g2, be2, m2, v2,
                     Wskip, out);
}
